// SPGCN_5583457484910
// MI455X (gfx1250) — compile-verified, matrix operand order checked by reading, not run on hardware
//
#include <hip/hip_runtime.h>

// ---------------------------------------------------------------------------
// SPGCN on gfx1250: 3x (GEMM[WMMA bf16 split] -> SpMM -> epilogue)
// ---------------------------------------------------------------------------

typedef __attribute__((ext_vector_type(16))) __bf16 v16bf;
typedef __attribute__((ext_vector_type(4)))  __bf16 v4bf;
typedef __attribute__((ext_vector_type(8)))  float  v8f;

#define N_NODES 50000
#define N_EDGES 1600000
#define BATCH   8
#define KDIM    128
#define M_ROWS  (BATCH * N_NODES)   // 400000

// ---------------------------------------------------------------------------
// row_ptr[i] = lower_bound(edge_rows, i); rows are pre-sorted by the harness.
// ---------------------------------------------------------------------------
__global__ void build_rowptr(const int* __restrict__ rows, int nE,
                             int* __restrict__ rowptr, int nNodes) {
    int i = blockIdx.x * blockDim.x + threadIdx.x;
    if (i > nNodes) return;
    int lo = 0, hi = nE;
    while (lo < hi) {
        int mid = (lo + hi) >> 1;
        if (rows[mid] < i) lo = mid + 1; else hi = mid;
    }
    rowptr[i] = lo;
}

// ---------------------------------------------------------------------------
// Pre-bake W into WMMA B-fragment layout, split bf16 hi/lo.
// Entry index: ((ntile*4 + kc)*32 + lane)*16 + e,  e = 2*v + p
//   k = kc*32 + (v<4?0:16) + (lane>>4)*8 + 2*(v&3) + p,  n = ntile*16 + (lane&15)
// Each wave later reads its fragment as one aligned 32B vector (2x b128).
// ---------------------------------------------------------------------------
__global__ void build_wfrag(const float* __restrict__ W,
                            __bf16* __restrict__ Fhi, __bf16* __restrict__ Flo,
                            int NOUT) {
    const int idx = blockIdx.x * blockDim.x + threadIdx.x;
    if (idx >= NOUT * KDIM) return;
    const int e    = idx & 15;
    const int lane = (idx >> 4) & 31;
    const int kc   = (idx >> 9) & 3;
    const int nt   = idx >> 11;
    const int v = e >> 1, p = e & 1;
    const int g = lane >> 4;
    const int k = kc * 32 + ((v & 4) << 2) + g * 8 + ((v & 3) << 1) + p;
    const int n = nt * 16 + (lane & 15);
    const float w = W[k * NOUT + n];
    const __bf16 h = (__bf16)w;
    Fhi[idx] = h;
    Flo[idx] = (__bf16)(w - (float)h);
}

// ---------------------------------------------------------------------------
// Dense GEMM: S[M,NOUT] = H[M,128] @ W[128,NOUT], split-bf16 WMMA (hi/lo).
// Block = 64-row M tile; NOUT/16 waves; each wave: one N tile x 4 M sub-tiles
// (B fragments reused across the 4 sub-tiles -> 48 WMMAs per wave).
// ---------------------------------------------------------------------------
template<int NOUT>
__global__ __launch_bounds__(256) void gemm_wmma(const float* __restrict__ H,
                                                 const __bf16* __restrict__ Fhi,
                                                 const __bf16* __restrict__ Flo,
                                                 float* __restrict__ S) {
    __shared__ __bf16 lds_hi[64][KDIM];
    __shared__ __bf16 lds_lo[64][KDIM];

    const int  tid = threadIdx.x;
    const long m0  = (long)blockIdx.x * 64;

    // Stage 64x128 A tile: f32 -> bf16 hi + residual lo. b128 global reads,
    // packed b64 LDS writes.
    const float4* H4 = (const float4*)(H + m0 * KDIM);
    for (int idx = tid; idx < 64 * KDIM / 4; idx += blockDim.x) {
        const float4 xv = H4[idx];
        const int r = idx >> 5;            // 32 float4 per 128-wide row
        const int k = (idx & 31) * 4;
        const float xs[4] = {xv.x, xv.y, xv.z, xv.w};
        v4bf h4, l4;
        #pragma unroll
        for (int i = 0; i < 4; ++i) {
            const __bf16 h = (__bf16)xs[i];
            h4[i] = h;
            l4[i] = (__bf16)(xs[i] - (float)h);
        }
        *(v4bf*)&lds_hi[r][k] = h4;
        *(v4bf*)&lds_lo[r][k] = l4;
    }
    __syncthreads();

    const int lane = tid & 31;
    const int wid  = tid >> 5;          // N tile of this wave
    const int g    = lane >> 4;         // lane half selects K phase
    const int rc   = lane & 15;         // A row / B column within tile

    v8f c[4] = {{}, {}, {}, {}};        // 4 M sub-tiles, f32 accumulators

    const v16bf* BH = (const v16bf*)Fhi;
    const v16bf* BL = (const v16bf*)Flo;

    #pragma unroll
    for (int kc = 0; kc < 4; ++kc) {     // K = 128 in chunks of 32
        const v16bf bhi = BH[(wid * 4 + kc) * 32 + lane];   // 2x b128, L2-hot
        const v16bf blo = BL[(wid * 4 + kc) * 32 + lane];
        const int kbase = kc * 32 + g * 8;
        #pragma unroll
        for (int mt = 0; mt < 4; ++mt) {
            const int row = mt * 16 + rc;
            v16bf ahi, alo;
            #pragma unroll
            for (int v = 0; v < 8; ++v) {
                const int k = kbase + ((v & 4) << 2) + ((v & 3) << 1);
                ahi[2 * v]     = lds_hi[row][k];
                ahi[2 * v + 1] = lds_hi[row][k + 1];
                alo[2 * v]     = lds_lo[row][k];
                alo[2 * v + 1] = lds_lo[row][k + 1];
            }
            // A*B ~= AhiBhi + AhiBlo + AloBhi (lo*lo ~ 2^-30, negligible)
            c[mt] = __builtin_amdgcn_wmma_f32_16x16x32_bf16(
                        false, ahi, false, bhi, (short)0, c[mt], false, false);
            c[mt] = __builtin_amdgcn_wmma_f32_16x16x32_bf16(
                        false, ahi, false, blo, (short)0, c[mt], false, false);
            c[mt] = __builtin_amdgcn_wmma_f32_16x16x32_bf16(
                        false, alo, false, bhi, (short)0, c[mt], false, false);
        }
    }

    // C/D layout: VGPR r -> M = r + 8*g, N = lane&15
    #pragma unroll
    for (int mt = 0; mt < 4; ++mt) {
        #pragma unroll
        for (int r = 0; r < 8; ++r) {
            const long row = m0 + mt * 16 + r + g * 8;
            S[row * NOUT + wid * 16 + rc] = c[mt][r];
        }
    }
}

// ---------------------------------------------------------------------------
// SpMM + epilogue. One wave per (batch, node); each lane owns D/32 columns.
// MODE 0: out = relu(agg + bias + resid)   (resid may alias out: each element
//         is read-then-written by exactly one wave, no cross-wave hazard)
// MODE 1: out = sigmoid(relu(agg + bias))
// ---------------------------------------------------------------------------
template<int D, int MODE>
__global__ __launch_bounds__(256) void spmm_kernel(const float* __restrict__ S,
                                                   const int* __restrict__ rowptr,
                                                   const int* __restrict__ cols,
                                                   const float* __restrict__ vals,
                                                   const float* __restrict__ bias,
                                                   const float* __restrict__ resid,
                                                   float* __restrict__ out) {
    constexpr int VW = D / 32;                 // floats per lane: 4 or 2
    const int  lane = threadIdx.x & 31;
    const int  wid  = threadIdx.x >> 5;
    const long gid  = (long)blockIdx.x * (blockDim.x >> 5) + wid;  // b*N + n
    const int  n    = (int)(gid % N_NODES);
    const int  b    = (int)(gid / N_NODES);

    const int start = rowptr[n];
    const int end   = rowptr[n + 1];
    const float* Sb = S + (long)b * N_NODES * D;

    float acc[VW];
    #pragma unroll
    for (int i = 0; i < VW; ++i) acc[i] = 0.0f;

    for (int e0 = start; e0 < end; e0 += 32) {
        const int cnt = (end - e0 < 32) ? (end - e0) : 32;
        int   cReg = 0;
        float vReg = 0.0f;
        if (lane < cnt) {
            cReg = cols[e0 + lane];
            vReg = vals[e0 + lane];
        }
        for (int j = 0; j < cnt; ++j) {
            const int   cj = __shfl(cReg, j);
            const float vj = __shfl(vReg, j);
            const float* src = Sb + (long)cj * D + lane * VW;
            if constexpr (VW == 4) {
                const float4 s4 = *(const float4*)src;   // 512B/line, L2-hot
                acc[0] += vj * s4.x; acc[1] += vj * s4.y;
                acc[2] += vj * s4.z; acc[3] += vj * s4.w;
            } else {
                const float2 s2 = *(const float2*)src;
                acc[0] += vj * s2.x; acc[1] += vj * s2.y;
            }
        }
    }

    const long obase = gid * D + lane * VW;
    #pragma unroll
    for (int i = 0; i < VW; ++i) {
        float v = acc[i] + bias[lane * VW + i];
        if constexpr (MODE == 0) v += resid[obase + i];
        v = v > 0.0f ? v : 0.0f;                 // relu
        if constexpr (MODE == 1) v = 1.0f / (1.0f + __expf(-v));
        out[obase + i] = v;
    }
}

// ---------------------------------------------------------------------------
// Orchestration. Inputs: x, edge_rows, edge_cols, edge_vals, W1,b1,W2,b2,W3,b3
// Workspace: rowptr | bufA (support) | bufB (h) | W-fragment planes (bf16)
// ---------------------------------------------------------------------------
extern "C" void kernel_launch(void* const* d_in, const int* in_sizes, int n_in,
                              void* d_out, int out_size, void* d_ws, size_t ws_size,
                              hipStream_t stream) {
    const float* x     = (const float*)d_in[0];
    const int*   erows = (const int*)  d_in[1];
    const int*   ecols = (const int*)  d_in[2];
    const float* evals = (const float*)d_in[3];
    const float* W1    = (const float*)d_in[4];
    const float* b1    = (const float*)d_in[5];
    const float* W2    = (const float*)d_in[6];
    const float* b2    = (const float*)d_in[7];
    const float* W3    = (const float*)d_in[8];
    const float* b3    = (const float*)d_in[9];
    float*       out   = (float*)d_out;

    char* ws = (char*)d_ws;
    int* rowptr = (int*)ws;
    size_t off = ((size_t)(N_NODES + 1) * sizeof(int) + 255) & ~(size_t)255;
    float* bufA = (float*)(ws + off);                       // support buffer
    float* bufB = bufA + (size_t)M_ROWS * KDIM;             // hidden state h
    __bf16* frag = (__bf16*)(bufB + (size_t)M_ROWS * KDIM);
    __bf16* F1h = frag;                 // 128x128 fragments: 16384 bf16 each
    __bf16* F1l = F1h + 16384;
    __bf16* F2h = F1l + 16384;
    __bf16* F2l = F2h + 16384;
    __bf16* F3h = F2l + 16384;          // 128x64 fragments: 8192 bf16 each
    __bf16* F3l = F3h + 8192;

    const int mtiles = M_ROWS / 64;                         // 6250
    const int sblk   = M_ROWS / 8;                          // 50000 (8 waves/blk)

    build_rowptr<<<(N_NODES + 256) / 256, 256, 0, stream>>>(erows, N_EDGES,
                                                            rowptr, N_NODES);
    build_wfrag<<<64, 256, 0, stream>>>(W1, F1h, F1l, 128);
    build_wfrag<<<64, 256, 0, stream>>>(W2, F2h, F2l, 128);
    build_wfrag<<<32, 256, 0, stream>>>(W3, F3h, F3l, 64);

    // Layer 1: h1 = relu(spmm(x@W1) + b1 + x)            -> bufB
    gemm_wmma<128><<<mtiles, 256, 0, stream>>>(x, F1h, F1l, bufA);
    spmm_kernel<128, 0><<<sblk, 256, 0, stream>>>(bufA, rowptr, ecols, evals,
                                                  b1, x, bufB);

    // Layer 2: h2 = relu(spmm(h1@W2) + b2 + h1)          -> bufB (in place)
    gemm_wmma<128><<<mtiles, 256, 0, stream>>>(bufB, F2h, F2l, bufA);
    spmm_kernel<128, 0><<<sblk, 256, 0, stream>>>(bufA, rowptr, ecols, evals,
                                                  b2, bufB, bufB);

    // Layer 3: out = sigmoid(relu(spmm(h2@W3) + b3))     -> d_out
    gemm_wmma<64><<<mtiles, 128, 0, stream>>>(bufB, F3h, F3l, bufA);
    spmm_kernel<64, 1><<<sblk, 256, 0, stream>>>(bufA, rowptr, ecols, evals,
                                                 b3, nullptr, out);
}
